// PointNet_9509057593717
// MI455X (gfx1250) — matrix operand order repaired and checked by reading, not compiled, not run on hardware
//
#include <hip/hip_runtime.h>
#include <hip/hip_bf16.h>

typedef __attribute__((ext_vector_type(16))) _Float16 v16h;
typedef __attribute__((ext_vector_type(8)))  _Float16 v8h;
typedef __attribute__((ext_vector_type(8)))  float    v8f;

#define BATCH 4
#define NPTS  8192
#define KNN   10
#define GF    100
#define MROWS (BATCH*NPTS)          // 32768 points total
#define LD_A1 32                    // layer1a out: 20 padded -> 32 (K of GEMM L2)
#define LD_X1 128                   // x1: 100 padded -> 128 (K of GEMM L3)
#define LD_X2 112                   // x2: 100 padded -> 112

// ---- workspace layout (bytes), total ~25.2 MiB ----
#define OFF_A1   0ull                               // f16 [MROWS][32]   2 MiB
#define OFF_X1   (OFF_A1  + (size_t)MROWS*LD_A1*2)  // f16 [MROWS][128]  8 MiB
#define OFF_X2   (OFF_X1  + (size_t)MROWS*LD_X1*2)  // f32 [MROWS][112] 14 MiB
#define OFF_BT1  (OFF_X2  + (size_t)MROWS*LD_X2*4)  // f16 [128][32]
#define OFF_BT2  (OFF_BT1 + (size_t)128*32*2)       // f16 [112][128]
#define OFF_POOL (OFF_BT2 + (size_t)112*128*2)      // f32 [4][100]

// -------- WMMA fragment loaders (f16, 16x16x32 tiles) --------
// A 16x32: lane holds row M=lane&15; K chunks [hs*8, hs*8+8) and [16+hs*8, ...)
__device__ __forceinline__ v16h load_fragA(const _Float16* __restrict__ p, int row,
                                           int ld, int k0, int lane) {
  const int hs = (lane >> 4) & 1;
  const _Float16* q = p + (size_t)row * ld + k0 + hs * 8;
  v8h lo = *(const v8h*)(q);
  v8h hi = *(const v8h*)(q + 16);
  v16h r;
#pragma unroll
  for (int i = 0; i < 8; ++i) { r[i] = lo[i]; r[i + 8] = hi[i]; }
  return r;
}
// B 32x16: lane holds column N=lane&15; K chunks [hs*8, hs*8+8) and [16+hs*8, ...)
// Weights stored transposed [out][in], so col indexes rows of Bt.
__device__ __forceinline__ v16h load_fragB(const _Float16* __restrict__ p, int col,
                                           int ld, int k0, int lane) {
  const int hs = (lane >> 4) & 1;
  const _Float16* q = p + (size_t)col * ld + k0 + hs * 8;
  v8h lo = *(const v8h*)(q);
  v8h hi = *(const v8h*)(q + 16);
  v16h r;
#pragma unroll
  for (int i = 0; i < 8; ++i) { r[i] = lo[i]; r[i + 8] = hi[i]; }
  return r;
}

// -------- weight prep: transpose + pad + cvt to f16; pad rows/cols are 0 --------
__global__ void __launch_bounds__(256)
prep_weights(const float* __restrict__ w1b, const float* __restrict__ w2,
             _Float16* __restrict__ Bt1, _Float16* __restrict__ Bt2) {
  int idx = blockIdx.x * 256 + threadIdx.x;
  if (idx < 128 * 32) {                 // Bt1[o][i] = w1b[i][o], (20->32, 100->128)
    int o = idx >> 5, i = idx & 31;
    float v = (o < GF && i < 20) ? w1b[i * GF + o] : 0.f;
    Bt1[idx] = (_Float16)v;
  }
  if (idx < 112 * 128) {                // Bt2[o][i] = w2[i][o], (100->128, 100->112)
    int o = idx >> 7, i = idx & 127;
    float v = (o < GF && i < GF) ? w2[i * GF + o] : 0.f;
    Bt2[idx] = (_Float16)v;
  }
}

// -------- fused pairwise-dist + top-10 (exact f32) + layer1a (13->20) --------
__global__ void __launch_bounds__(256)
knn_l1_kernel(const float* __restrict__ x, const float* __restrict__ w1a,
              const float* __restrict__ b1a, _Float16* __restrict__ A1) {
  __shared__ float tile[256 * 3];
  const int tid = threadIdx.x;
  const int b = blockIdx.x >> 5;                      // 32 blocks per batch
  const int n = ((blockIdx.x & 31) << 8) + tid;
  const float* xb = x + (size_t)b * NPTS * 3;
  const float px = xb[n * 3 + 0], py = xb[n * 3 + 1], pz = xb[n * 3 + 2];

  float best[KNN];
#pragma unroll
  for (int q = 0; q < KNN; ++q) best[q] = 3.4e38f;

  for (int t0 = 0; t0 < NPTS; t0 += 256) {
    __syncthreads();
    tile[tid * 3 + 0] = xb[(t0 + tid) * 3 + 0];
    tile[tid * 3 + 1] = xb[(t0 + tid) * 3 + 1];
    tile[tid * 3 + 2] = xb[(t0 + tid) * 3 + 2];
    __syncthreads();
#pragma unroll 4
    for (int j = 0; j < 256; ++j) {
      float dx = px - tile[j * 3 + 0];
      float dy = py - tile[j * 3 + 1];
      float dz = pz - tile[j * 3 + 2];
      float d = fmaf(dz, dz, fmaf(dy, dy, dx * dx));
      if (d < best[KNN - 1]) {          // rare path: ascending bubble insert
        float v = d;
#pragma unroll
        for (int q = 0; q < KNN; ++q) {
          float c = best[q];
          best[q] = fminf(c, v);
          v = fmaxf(c, v);
        }
      }
    }
  }

  float h[13];
  h[0] = px; h[1] = py; h[2] = pz;
#pragma unroll
  for (int q = 0; q < KNN; ++q) h[3 + q] = best[q];

  _Float16* arow = A1 + (size_t)(b * NPTS + n) * LD_A1;
#pragma unroll
  for (int o = 0; o < 20; ++o) {        // layer1a: relu(h @ w1a + b1a)
    float acc = b1a[o];
#pragma unroll
    for (int i = 0; i < 13; ++i) acc = fmaf(h[i], w1a[i * 20 + o], acc);
    arow[o] = (_Float16)fmaxf(acc, 0.f);
  }
#pragma unroll
  for (int o = 20; o < 32; ++o) arow[o] = (_Float16)0.f;   // K padding
}

// -------- GEMM L2: x1 = relu(a1 @ w1b + b1b), M=32768 K=32 N=128(pad) --------
__global__ void __launch_bounds__(256)
gemm_l2_wmma(const _Float16* __restrict__ A1, const _Float16* __restrict__ Bt1,
             const float* __restrict__ b1b, _Float16* __restrict__ X1) {
  const int lane = threadIdx.x & 31;
  const int wave = (blockIdx.x << 3) + (threadIdx.x >> 5);   // one M-tile per wave
  const int m0 = wave << 4;
  const int rbase = m0 + ((lane >> 4) << 3);

  v16h a = load_fragA(A1, m0 + (lane & 15), LD_A1, 0, lane);

#pragma unroll
  for (int nt = 0; nt < 8; ++nt) {
    const int col = (nt << 4) + (lane & 15);
    v16h bf = load_fragB(Bt1, col, 32, 0, lane);
    v8f c = {};
    c = __builtin_amdgcn_wmma_f32_16x16x32_f16(false, a, false, bf,
                                               (short)0, c, false, false);
    float bias = (col < GF) ? b1b[col] : 0.f;
#pragma unroll
    for (int r = 0; r < 8; ++r) {
      float v = fmaxf(c[r] + bias, 0.f);
      X1[(size_t)(rbase + r) * LD_X1 + col] = (_Float16)v;
    }
  }
}

// -------- GEMM L3: x2 = relu(x1 @ w2 + b2), M=32768 K=128(pad) N=112(pad) ----
__global__ void __launch_bounds__(256)
gemm_l3_wmma(const _Float16* __restrict__ X1, const _Float16* __restrict__ Bt2,
             const float* __restrict__ b2, float* __restrict__ X2) {
  const int lane = threadIdx.x & 31;
  const int wave = (blockIdx.x << 3) + (threadIdx.x >> 5);
  const int m0 = wave << 4;
  const int mr = m0 + (lane & 15);
  const int rbase = m0 + ((lane >> 4) << 3);

  v16h a0 = load_fragA(X1, mr, LD_X1, 0,  lane);   // A frags register-resident
  v16h a1 = load_fragA(X1, mr, LD_X1, 32, lane);
  v16h a2 = load_fragA(X1, mr, LD_X1, 64, lane);
  v16h a3 = load_fragA(X1, mr, LD_X1, 96, lane);

#pragma unroll
  for (int nt = 0; nt < 7; ++nt) {
    const int col = (nt << 4) + (lane & 15);
    v8f c = {};
    c = __builtin_amdgcn_wmma_f32_16x16x32_f16(false, a0, false,
          load_fragB(Bt2, col, 128, 0,  lane), (short)0, c, false, false);
    c = __builtin_amdgcn_wmma_f32_16x16x32_f16(false, a1, false,
          load_fragB(Bt2, col, 128, 32, lane), (short)0, c, false, false);
    c = __builtin_amdgcn_wmma_f32_16x16x32_f16(false, a2, false,
          load_fragB(Bt2, col, 128, 64, lane), (short)0, c, false, false);
    c = __builtin_amdgcn_wmma_f32_16x16x32_f16(false, a3, false,
          load_fragB(Bt2, col, 128, 96, lane), (short)0, c, false, false);
    float bias = (col < GF) ? b2[col] : 0.f;
#pragma unroll
    for (int r = 0; r < 8; ++r)
      X2[(size_t)(rbase + r) * LD_X2 + col] = fmaxf(c[r] + bias, 0.f);
  }
}

// -------- deterministic mean pool over N --------
__global__ void __launch_bounds__(256)
pool_kernel(const float* __restrict__ X2, float* __restrict__ pool) {
  __shared__ float red[256];
  const int b = blockIdx.x / GF;
  const int g = blockIdx.x % GF;
  float s = 0.f;
  for (int i = threadIdx.x; i < NPTS; i += 256)
    s += X2[(size_t)(b * NPTS + i) * LD_X2 + g];
  red[threadIdx.x] = s;
  __syncthreads();
  for (int st = 128; st > 0; st >>= 1) {
    if (threadIdx.x < st) red[threadIdx.x] += red[threadIdx.x + st];
    __syncthreads();
  }
  if (threadIdx.x == 0) pool[b * GF + g] = red[0] * (1.f / NPTS);
}

// -------- head: [x1|pool](200) -> 20 -> 10 -> 2 + log_softmax --------
__global__ void __launch_bounds__(256)
head_kernel(const _Float16* __restrict__ X1, const float* __restrict__ pool,
            const float* __restrict__ w3a, const float* __restrict__ b3a,
            const float* __restrict__ w3b, const float* __restrict__ b3b,
            const float* __restrict__ w3c, const float* __restrict__ b3c,
            float* __restrict__ out) {
  const int idx = blockIdx.x * 256 + threadIdx.x;     // 0..MROWS-1
  const int b = idx >> 13;                            // idx / NPTS
  const _Float16* xr = X1 + (size_t)idx * LD_X1;
  const float* pr = pool + b * GF;

  float acc[20];
#pragma unroll
  for (int o = 0; o < 20; ++o) acc[o] = b3a[o];
  for (int i = 0; i < GF; ++i) {
    float v = (float)xr[i];
#pragma unroll
    for (int o = 0; o < 20; ++o) acc[o] = fmaf(v, w3a[i * 20 + o], acc[o]);
  }
  for (int i = 0; i < GF; ++i) {
    float v = pr[i];
#pragma unroll
    for (int o = 0; o < 20; ++o) acc[o] = fmaf(v, w3a[(GF + i) * 20 + o], acc[o]);
  }
  float o1[20];
#pragma unroll
  for (int o = 0; o < 20; ++o) o1[o] = fmaxf(acc[o], 0.f);

  float o2[10];
#pragma unroll
  for (int p = 0; p < 10; ++p) {
    float a = b3b[p];
#pragma unroll
    for (int o = 0; o < 20; ++o) a = fmaf(o1[o], w3b[o * 10 + p], a);
    o2[p] = fmaxf(a, 0.f);
  }
  float z0 = b3c[0], z1 = b3c[1];
#pragma unroll
  for (int p = 0; p < 10; ++p) {
    z0 = fmaf(o2[p], w3c[p * 2 + 0], z0);
    z1 = fmaf(o2[p], w3c[p * 2 + 1], z1);
  }
  float m = fmaxf(z0, z1);
  float lse = logf(expf(z0 - m) + expf(z1 - m)) + m;
  out[(size_t)idx * 2 + 0] = z0 - lse;
  out[(size_t)idx * 2 + 1] = z1 - lse;
}

extern "C" void kernel_launch(void* const* d_in, const int* in_sizes, int n_in,
                              void* d_out, int out_size, void* d_ws, size_t ws_size,
                              hipStream_t stream) {
  const float* x   = (const float*)d_in[0];
  const float* w1a = (const float*)d_in[1];
  const float* b1a = (const float*)d_in[2];
  const float* w1b = (const float*)d_in[3];
  const float* b1b = (const float*)d_in[4];
  const float* w2  = (const float*)d_in[5];
  const float* b2  = (const float*)d_in[6];
  const float* w3a = (const float*)d_in[7];
  const float* b3a = (const float*)d_in[8];
  const float* w3b = (const float*)d_in[9];
  const float* b3b = (const float*)d_in[10];
  const float* w3c = (const float*)d_in[11];
  const float* b3c = (const float*)d_in[12];

  char* ws = (char*)d_ws;
  _Float16* A1   = (_Float16*)(ws + OFF_A1);
  _Float16* X1   = (_Float16*)(ws + OFF_X1);
  float*    X2   = (float*)   (ws + OFF_X2);
  _Float16* Bt1  = (_Float16*)(ws + OFF_BT1);
  _Float16* Bt2  = (_Float16*)(ws + OFF_BT2);
  float*    pl   = (float*)   (ws + OFF_POOL);

  prep_weights<<<56, 256, 0, stream>>>(w1b, w2, Bt1, Bt2);
  knn_l1_kernel<<<BATCH * NPTS / 256, 256, 0, stream>>>(x, w1a, b1a, A1);
  gemm_l2_wmma<<<MROWS / 16 / 8, 256, 0, stream>>>(A1, Bt1, b1b, X1);
  gemm_l3_wmma<<<MROWS / 16 / 8, 256, 0, stream>>>(X1, Bt2, b2, X2);
  pool_kernel<<<BATCH * GF, 256, 0, stream>>>(X2, pl);
  head_kernel<<<MROWS / 256, 256, 0, stream>>>(X1, pl, w3a, b3a, w3b, b3b,
                                               w3c, b3c, (float*)d_out);
}
